// GroupRNN_64656437674083
// MI455X (gfx1250) — matrix operand order repaired and checked by reading, not compiled, not run on hardware
//
#include <hip/hip_runtime.h>
#include <hip/hip_bf16.h>

typedef __attribute__((ext_vector_type(16))) __bf16 v16bf;
typedef __attribute__((ext_vector_type(8)))  __bf16 v8bf;
typedef __attribute__((ext_vector_type(8)))  float  v8f;

#define GRP   4
#define HID   64
#define BATCH 32
#define SEQ   1000
#define FEAT  256              // 4H = G*H
#define M_TOT (BATCH*SEQ)      // 32000 rows

__device__ __forceinline__ float sigmoidf_(float x) { return 1.0f / (1.0f + __expf(-x)); }

// ---------------------------------------------------------------------------
// fp32 -> bf16 weight conversion (done once per launch, tiny)
// ---------------------------------------------------------------------------
__global__ void cvt_bf16_kernel(const float* __restrict__ src, __bf16* __restrict__ dst, int n) {
    int i = blockIdx.x * blockDim.x + threadIdx.x;
    if (i < n) dst[i] = (__bf16)src[i];
}

// ---------------------------------------------------------------------------
// Input projection: xp[g,m,:] = src_rows(m, cols g*64..g*64+63) @ W_ihᵀ + bias
// Optional fused BN (scale/shift per source feature) for layer 2.
// Block = 256 threads (8 waves); block handles one (g, 16-row m-tile);
// each wave computes 2 of the 16 N-tiles with 2 WMMA (K=64) each.
// ---------------------------------------------------------------------------
__global__ __launch_bounds__(256) void proj_kernel(
    const float* __restrict__ src,    // (M_TOT, 256) row-major
    const __bf16* __restrict__ wih,   // (G, 256, 64) bf16 row-major
    const float* __restrict__ bias,   // (G, 256)
    const float* __restrict__ scale,  // (256) fused BN scale (or unused)
    const float* __restrict__ shift,  // (256) fused BN shift (or unused)
    int use_bn,
    float* __restrict__ xp)           // (G, M_TOT, 256)
{
    int blk = blockIdx.x;
    int g   = blk / (M_TOT / 16);
    int mt  = blk % (M_TOT / 16);
    int m0  = mt * 16;

    int lane = threadIdx.x & 31;
    int wv   = threadIdx.x >> 5;     // 0..7
    int r    = lane & 15;
    int hk   = lane >> 4;            // 0/1: K half-select per ISA A/B layouts

    // ---- A tiles: 16 rows x K=64 of source, bf16, per-ISA lane layout ----
    v16bf a[2];
    const float* sp = src + (size_t)(m0 + r) * FEAT + g * HID;
#pragma unroll
    for (int ks = 0; ks < 2; ++ks) {
        int kb = ks * 32 + hk * 8;
        v8f lo = *(const v8f*)(sp + kb);
        v8f hi = *(const v8f*)(sp + kb + 16);
        if (use_bn) {
#pragma unroll
            for (int e = 0; e < 8; ++e) {
                int f0 = g * HID + kb + e;
                int f1 = f0 + 16;
                lo[e] = lo[e] * scale[f0] + shift[f0];
                hi[e] = hi[e] * scale[f1] + shift[f1];
            }
        }
        v16bf av;
#pragma unroll
        for (int e = 0; e < 8; ++e) { av[e] = (__bf16)lo[e]; av[e + 8] = (__bf16)hi[e]; }
        a[ks] = av;
    }

    float* dst = xp + ((size_t)g * M_TOT + m0 + 8 * hk) * FEAT;

    // ---- two N tiles per wave; bias folded into C init ----
#pragma unroll
    for (int q = 0; q < 2; ++q) {
        int n0 = (wv + q * 8) * 16;
        int n  = n0 + r;                       // output gate column
        float bv = bias[g * FEAT + n];
        v8f acc;
#pragma unroll
        for (int i = 0; i < 8; ++i) acc[i] = bv;
#pragma unroll
        for (int ks = 0; ks < 2; ++ks) {
            // B column n = W row n: two 8-bf16 runs at k = 8h.. and 16+8h..
            const __bf16* wq = wih + ((size_t)g * FEAT + n) * HID + ks * 32 + hk * 8;
            v16bf bm = __builtin_shufflevector(*(const v8bf*)wq, *(const v8bf*)(wq + 16),
                0, 1, 2, 3, 4, 5, 6, 7, 8, 9, 10, 11, 12, 13, 14, 15);
            acc = __builtin_amdgcn_wmma_f32_16x16x32_bf16(
                false, a[ks], false, bm, (short)0, acc, false, false);
        }
        // coalesced row-major store: row m0 + i + 8*hk, col n
#pragma unroll
        for (int i = 0; i < 8; ++i)
            dst[(size_t)i * FEAT + n] = acc[i];
    }
}

// ---------------------------------------------------------------------------
// Persistent recurrent kernel: one workgroup per group, 256 threads (8 waves).
// Wave w owns (jt = w&3, mt = w>>2) and computes ALL FOUR gate tiles for its
// cells, so i/f/g/o for cell (b,j) land in the same lane & VGPR slot of the
// four accumulators: the LSTM cell update runs entirely in registers.
// - W_hh B-tiles: hoisted out of the t-loop (8 x v16bf, loaded once).
// - c state: 8 VGPRs per lane for the whole sequence.
// - h: double-buffered in LDS (2 x 4KB) -> ONE barrier per step.
// - xp reads: software-pipelined one step ahead in registers (xnext), so the
//   only per-step serial chain is ds_load(h) -> WMMA -> trans -> ds_store(h).
// ---------------------------------------------------------------------------
__global__ __launch_bounds__(256) void lstm_recur_kernel(
    const __bf16* __restrict__ whh,   // (G, 256, 64) bf16
    const float* __restrict__ xp,     // (G, M_TOT, 256)
    float* __restrict__ y,            // (M_TOT, 256) output features
    int interleave)                   // 1: f = j*G+g (layer1), 0: f = g*H+j (layer2)
{
    __shared__ __bf16 hbuf[2][BATCH * HID];   // 2 x 4KB ping-pong hidden state

    int g    = blockIdx.x;
    int tid  = threadIdx.x;
    int lane = tid & 31;
    int wv   = tid >> 5;          // 0..7
    int jt   = wv & 3;            // hidden-unit tile (16 j's)
    int mt   = wv >> 2;           // batch-row tile (0/1)
    int r    = lane & 15;
    int hk   = lane >> 4;
    int j    = jt * 16 + r;

    // ---- loop-invariant W_hh B-tiles: 4 gates x 2 K-steps ----
    v16bf bm[4][2];
#pragma unroll
    for (int gate = 0; gate < 4; ++gate) {
        const __bf16* wp = whh + ((size_t)g * FEAT + gate * HID + j) * HID + hk * 8;
#pragma unroll
        for (int ks = 0; ks < 2; ++ks)
            bm[gate][ks] = __builtin_shufflevector(*(const v8bf*)(wp + ks * 32), *(const v8bf*)(wp + ks * 32 + 16),
                0, 1, 2, 3, 4, 5, 6, 7, 8, 9, 10, 11, 12, 13, 14, 15);
    }

    for (int i = tid; i < BATCH * HID; i += 256) hbuf[0][i] = (__bf16)0.0f;
    v8f c;
#pragma unroll
    for (int i = 0; i < 8; ++i) c[i] = 0.0f;
    __syncthreads();

    const float* xg = xp + (size_t)g * M_TOT * FEAT;     // group base
    // cooperative prefetch: thread covers row (tid&31), 128B chunk (tid>>5)
    const float* pf_base = xg + (size_t)(tid & 31) * SEQ * FEAT + (tid >> 5) * 32;

    const int ff = interleave ? (j * GRP + g) : (g * HID + j);
    float* yb = y + ff;          // + (b*SEQ + t)*FEAT per cell

    // ---- preload xp for t = 0 into registers ----
    v8f xnext[4];
#pragma unroll
    for (int gate = 0; gate < 4; ++gate) {
        int n = gate * HID + j;
#pragma unroll
        for (int i = 0; i < 8; ++i)
            xnext[gate][i] = xg[(size_t)(mt * 16 + i + 8 * hk) * (SEQ * FEAT) + n];
    }

    int cur = 0;
    for (int t = 0; t < SEQ; ++t) {
        if (t + 2 < SEQ)
            __builtin_prefetch(pf_base + (size_t)(t + 2) * FEAT, 0, 0);

        // consume pipelined C init, then immediately issue loads for t+1
        // (independent of h -> overlaps the whole step's compute + barrier)
        v8f acc[4];
#pragma unroll
        for (int gate = 0; gate < 4; ++gate) acc[gate] = xnext[gate];

        if (t + 1 < SEQ) {
            const float* xt1 = xg + (size_t)(t + 1) * FEAT;
#pragma unroll
            for (int gate = 0; gate < 4; ++gate) {
                int n = gate * HID + j;
#pragma unroll
                for (int i = 0; i < 8; ++i)
                    xnext[gate][i] = xt1[(size_t)(mt * 16 + i + 8 * hk) * (SEQ * FEAT) + n];
            }
        }

        // ---- A tile (h of previous step) for this wave's batch rows ----
        const __bf16* hc = hbuf[cur];
        v16bf a[2];
#pragma unroll
        for (int ks = 0; ks < 2; ++ks) {
            const __bf16* hp = hc + (size_t)(mt * 16 + r) * HID + ks * 32 + hk * 8;
            v8bf lo = *(const v8bf*)hp;
            v8bf hi = *(const v8bf*)(hp + 16);
            a[ks] = __builtin_shufflevector(lo, hi,
                0, 1, 2, 3, 4, 5, 6, 7, 8, 9, 10, 11, 12, 13, 14, 15);
        }

        // ---- 4 gate tiles ----
#pragma unroll
        for (int gate = 0; gate < 4; ++gate)
#pragma unroll
            for (int ks = 0; ks < 2; ++ks)
                acc[gate] = __builtin_amdgcn_wmma_f32_16x16x32_bf16(
                    false, a[ks], false, bm[gate][ks], (short)0, acc[gate], false, false);

        // ---- LSTM cell update, fully in registers ----
        __bf16* hn = hbuf[cur ^ 1];
#pragma unroll
        for (int i = 0; i < 8; ++i) {
            int b = mt * 16 + i + 8 * hk;
            float ig = sigmoidf_(acc[0][i]);
            float fg = sigmoidf_(acc[1][i]);
            float gg = tanhf(acc[2][i]);
            float og = sigmoidf_(acc[3][i]);
            float cc = fg * c[i] + ig * gg;
            c[i] = cc;
            float h = og * tanhf(cc);
            hn[b * HID + j] = (__bf16)h;
            yb[((size_t)b * SEQ + t) * FEAT] = h;
        }
        __syncthreads();
        cur ^= 1;
    }
}

// ---------------------------------------------------------------------------
// BatchNorm: per-feature mean/var over (B,T), folded into scale/shift
// ---------------------------------------------------------------------------
__global__ __launch_bounds__(256) void bnstats_kernel(
    const float* __restrict__ y, const float* __restrict__ gamma,
    const float* __restrict__ beta, float* __restrict__ scale, float* __restrict__ shift)
{
    __shared__ float rs[256], rq[256];
    int f = blockIdx.x, tid = threadIdx.x;
    float s = 0.f, q = 0.f;
    for (int m = tid; m < M_TOT; m += 256) {
        float v = y[(size_t)m * FEAT + f];
        s += v; q += v * v;
    }
    rs[tid] = s; rq[tid] = q;
    __syncthreads();
    for (int o = 128; o > 0; o >>= 1) {
        if (tid < o) { rs[tid] += rs[tid + o]; rq[tid] += rq[tid + o]; }
        __syncthreads();
    }
    if (tid == 0) {
        float mean = rs[0] * (1.0f / M_TOT);
        float var  = rq[0] * (1.0f / M_TOT) - mean * mean;
        float sc = gamma[f] * rsqrtf(var + 1e-5f);
        scale[f] = sc;
        shift[f] = beta[f] - mean * sc;
    }
}

__global__ void bn_apply_kernel(const float* __restrict__ y, const float* __restrict__ scale,
                                const float* __restrict__ shift, float* __restrict__ out)
{
    size_t i = (size_t)blockIdx.x * blockDim.x + threadIdx.x;
    if (i < (size_t)M_TOT * FEAT) {
        int f = (int)(i & (FEAT - 1));
        out[i] = y[i] * scale[f] + shift[f];
    }
}

// ---------------------------------------------------------------------------
extern "C" void kernel_launch(void* const* d_in, const int* in_sizes, int n_in,
                              void* d_out, int out_size, void* d_ws, size_t ws_size,
                              hipStream_t stream) {
    (void)in_sizes; (void)n_in; (void)out_size; (void)ws_size;
    const float* inpt   = (const float*)d_in[0];
    const float* Wih1   = (const float*)d_in[1];
    const float* Whh1   = (const float*)d_in[2];
    const float* b1     = (const float*)d_in[3];
    const float* Wih2   = (const float*)d_in[4];
    const float* Whh2   = (const float*)d_in[5];
    const float* b2     = (const float*)d_in[6];
    const float* gamma1 = (const float*)d_in[7];
    const float* beta1  = (const float*)d_in[8];
    const float* gamma2 = (const float*)d_in[9];
    const float* beta2  = (const float*)d_in[10];
    float* out = (float*)d_out;

    char* ws = (char*)d_ws;
    size_t off = 0;
    float*  xp    = (float*)(ws + off);  off += (size_t)GRP * M_TOT * FEAT * 4; // reused by both layers
    float*  y1    = (float*)(ws + off);  off += (size_t)M_TOT * FEAT * 4;
    float*  y2    = (float*)(ws + off);  off += (size_t)M_TOT * FEAT * 4;
    __bf16* wih1b = (__bf16*)(ws + off); off += (size_t)GRP * FEAT * HID * 2;
    __bf16* whh1b = (__bf16*)(ws + off); off += (size_t)GRP * FEAT * HID * 2;
    __bf16* wih2b = (__bf16*)(ws + off); off += (size_t)GRP * FEAT * HID * 2;
    __bf16* whh2b = (__bf16*)(ws + off); off += (size_t)GRP * FEAT * HID * 2;
    float*  scale1 = (float*)(ws + off); off += FEAT * 4;
    float*  shift1 = (float*)(ws + off); off += FEAT * 4;
    float*  scale2 = (float*)(ws + off); off += FEAT * 4;
    float*  shift2 = (float*)(ws + off); off += FEAT * 4;

    const int wn = GRP * FEAT * HID;
    cvt_bf16_kernel<<<(wn + 255) / 256, 256, 0, stream>>>(Wih1, wih1b, wn);
    cvt_bf16_kernel<<<(wn + 255) / 256, 256, 0, stream>>>(Whh1, whh1b, wn);
    cvt_bf16_kernel<<<(wn + 255) / 256, 256, 0, stream>>>(Wih2, wih2b, wn);
    cvt_bf16_kernel<<<(wn + 255) / 256, 256, 0, stream>>>(Whh2, whh2b, wn);

    // Layer 1
    proj_kernel<<<GRP * (M_TOT / 16), 256, 0, stream>>>(inpt, wih1b, b1, nullptr, nullptr, 0, xp);
    lstm_recur_kernel<<<GRP, 256, 0, stream>>>(whh1b, xp, y1, 1);
    bnstats_kernel<<<FEAT, 256, 0, stream>>>(y1, gamma1, beta1, scale1, shift1);

    // Layer 2 (BN1 fused into projection A-load)
    proj_kernel<<<GRP * (M_TOT / 16), 256, 0, stream>>>(y1, wih2b, b2, scale1, shift1, 1, xp);
    lstm_recur_kernel<<<GRP, 256, 0, stream>>>(whh2b, xp, y2, 0);
    bnstats_kernel<<<FEAT, 256, 0, stream>>>(y2, gamma2, beta2, scale2, shift2);

    const size_t ntot = (size_t)M_TOT * FEAT;
    bn_apply_kernel<<<(int)((ntot + 255) / 256), 256, 0, stream>>>(y2, scale2, shift2, out);
}
